// GCN_res_86354612454001
// MI455X (gfx1250) — matrix operand order repaired and checked by reading, not compiled, not run on hardware
//
#include <hip/hip_runtime.h>
#include <hip/hip_bf16.h>

typedef __bf16 bf16_t;
typedef __attribute__((ext_vector_type(16))) __bf16 v16bf;
typedef __attribute__((ext_vector_type(8)))  float  v8f;

#define NN 100000
#define NE 3200000
#define MT 6250   // NN/16 M-tiles (exact)

// ---------------- fp32 -> (bf16 hi, bf16 lo) split ----------------
__global__ void k_split(const float* __restrict__ in, bf16_t* __restrict__ hi,
                        bf16_t* __restrict__ lo, int n) {
  int i = blockIdx.x * blockDim.x + threadIdx.x;
  if (i >= n) return;
  float x = in[i];
  bf16_t h = (bf16_t)x;
  hi[i] = h;
  lo[i] = (bf16_t)(x - (float)h);
}

// ---- transpose + split + pad: W[K x dout] -> Wt_hi/lo[doutPad x K] ----
__global__ void k_wsplit(const float* __restrict__ W, bf16_t* __restrict__ hi,
                         bf16_t* __restrict__ lo, int K, int dout, int doutPad) {
  int i = blockIdx.x * blockDim.x + threadIdx.x;
  if (i >= doutPad * K) return;
  int n = i / K;
  int k = i - n * K;
  float x = (n < dout) ? W[k * dout + n] : 0.0f;
  bf16_t h = (bf16_t)x;
  hi[i] = h;
  lo[i] = (bf16_t)(x - (float)h);
}

// ---------------- WMMA GEMM: S[M x ldS] = A[M x K] * Wt^T ----------------
// bf16x3 split (Ahi*Bhi + Ahi*Blo + Alo*Bhi, f32 accumulate) ~ fp32 accuracy.
// WMMAs are issued round-robin over a group of 4 n-tiles so dependent WMMAs
// on the same accumulator are >=3 instructions apart (bf16 WMMA->WMMA RAW
// hazard needs ~5 slots; the scheduler fills the rest with loads/VALU).
template <int NT>
__global__ __launch_bounds__(256) void k_gemm(
    const bf16_t* __restrict__ Ahi, const bf16_t* __restrict__ Alo,
    const bf16_t* __restrict__ Bhi, const bf16_t* __restrict__ Blo,
    float* __restrict__ S, int K, int ldS, int Mtiles) {
  const int lane = threadIdx.x & 31;
  const int wid  = threadIdx.x >> 5;
  const int mtile = blockIdx.y * 8 + wid;
  if (mtile >= Mtiles) return;                 // uniform per-wave exit
  const int half = lane >> 4;
  const int l16  = lane & 15;
  const int row  = mtile * 16 + l16;
  const int ncol0 = blockIdx.x * (NT * 16);

  v8f acc[NT];
#pragma unroll
  for (int t = 0; t < NT; ++t) acc[t] = (v8f){0.f,0.f,0.f,0.f,0.f,0.f,0.f,0.f};

  union Frag { v16bf v; uint4 u[2]; };

  for (int k0 = 0; k0 < K; k0 += 32) {
    // A fragment: lane(row=l16, half) holds K = half*8+{0..7} and 16+half*8+{0..7}
    Frag ah, al;
    const size_t ab = (size_t)row * K + k0 + half * 8;
    ah.u[0] = *(const uint4*)(Ahi + ab);
    ah.u[1] = *(const uint4*)(Ahi + ab + 16);
    al.u[0] = *(const uint4*)(Alo + ab);
    al.u[1] = *(const uint4*)(Alo + ab + 16);
#pragma unroll
    for (int t0 = 0; t0 < NT; t0 += 4) {
      // B fragments: lane(col=l16, half) holds K = half*8+{0..7} and 16+half*8+{0..7}
      Frag bh[4], bl[4];
#pragma unroll
      for (int g = 0; g < 4; ++g) {
        if (t0 + g < NT) {
          const size_t bb = (size_t)(ncol0 + (t0 + g) * 16 + l16) * K + k0 + half * 8;
          bh[g].u[0] = *(const uint4*)(Bhi + bb);
          bh[g].u[1] = *(const uint4*)(Bhi + bb + 16);
          bl[g].u[0] = *(const uint4*)(Blo + bb);
          bl[g].u[1] = *(const uint4*)(Blo + bb + 16);
        }
      }
#pragma unroll
      for (int g = 0; g < 4; ++g)
        if (t0 + g < NT)
          acc[t0+g] = __builtin_amdgcn_wmma_f32_16x16x32_bf16(false, ah.v, false, bh[g].v,
                                                              (short)0, acc[t0+g], false, false);
#pragma unroll
      for (int g = 0; g < 4; ++g)
        if (t0 + g < NT)
          acc[t0+g] = __builtin_amdgcn_wmma_f32_16x16x32_bf16(false, ah.v, false, bl[g].v,
                                                              (short)0, acc[t0+g], false, false);
#pragma unroll
      for (int g = 0; g < 4; ++g)
        if (t0 + g < NT)
          acc[t0+g] = __builtin_amdgcn_wmma_f32_16x16x32_bf16(false, al.v, false, bh[g].v,
                                                              (short)0, acc[t0+g], false, false);
    }
  }
  // C/D layout: VGPR i, lane -> row = i + half*8, col = l16
#pragma unroll
  for (int t = 0; t < NT; ++t) {
    const int c = ncol0 + t * 16 + l16;
#pragma unroll
    for (int i = 0; i < 8; ++i) {
      S[(size_t)(mtile * 16 + half * 8 + i) * ldS + c] = acc[t][i];
    }
  }
}

// ---------------- edge scatter: agg[dst] += val * S[src] ----------------
// thread -> (edge, 4-feature chunk); consecutive lanes cover consecutive
// features of one edge -> coalesced gather + coalesced hardware f32 atomics.
template <int CHUNKS, int LD>
__global__ void k_scatter(const float* __restrict__ S, const int* __restrict__ src,
                          const int* __restrict__ dst, const float* __restrict__ val,
                          float* __restrict__ agg, unsigned total) {
  unsigned i = blockIdx.x * blockDim.x + threadIdx.x;
  if (i >= total) return;
  unsigned e = i / (unsigned)CHUNKS;
  unsigned c = i - e * (unsigned)CHUNKS;
  int s = src[e], d = dst[e];
  float w = val[e];
  const float4 sv = *(const float4*)(S + (size_t)s * LD + c * 4);
  float* ap = agg + (size_t)d * LD + c * 4;
  unsafeAtomicAdd(ap + 0, w * sv.x);   // -> global_atomic_add_f32
  unsafeAtomicAdd(ap + 1, w * sv.y);
  unsafeAtomicAdd(ap + 2, w * sv.z);
  unsafeAtomicAdd(ap + 3, w * sv.w);
}

// ---------------- epilogues (dout is power of two -> mask) ----------------
__global__ void k_bias_relu(const float* __restrict__ agg, const float* __restrict__ b,
                            float* __restrict__ out, int mask, int n) {
  int i = blockIdx.x * blockDim.x + threadIdx.x;
  if (i >= n) return;
  out[i] = fmaxf(agg[i] + b[i & mask], 0.0f);
}

// id2 = relu(agg+b); h = id2 + h_old(id1)   (layer 2)
__global__ void k_bias_relu_save_addres(const float* __restrict__ agg,
                                        const float* __restrict__ b,
                                        float* __restrict__ id2,
                                        float* __restrict__ h, int mask, int n) {
  int i = blockIdx.x * blockDim.x + threadIdx.x;
  if (i >= n) return;
  float r = fmaxf(agg[i] + b[i & mask], 0.0f);
  id2[i] = r;
  h[i] = r + h[i];
}

// h = relu(agg+b) + id2   (layer 3)
__global__ void k_bias_relu_add(const float* __restrict__ agg, const float* __restrict__ b,
                                const float* __restrict__ id2, float* __restrict__ h,
                                int mask, int n) {
  int i = blockIdx.x * blockDim.x + threadIdx.x;
  if (i >= n) return;
  h[i] = fmaxf(agg[i] + b[i & mask], 0.0f) + id2[i];
}

// relu(agg+b5) then row log-softmax over 40 classes (agg stride 48). wave/row.
__global__ __launch_bounds__(256) void k_logsoftmax(const float* __restrict__ agg,
                                                    const float* __restrict__ b,
                                                    float* __restrict__ out, int rows) {
  int lane = threadIdx.x & 31;
  int wid  = threadIdx.x >> 5;
  int row  = blockIdx.x * 8 + wid;
  if (row >= rows) return;
  const float* a = agg + (size_t)row * 48;
  int c0 = lane, c1 = lane + 32;
  bool p0 = c0 < 40, p1 = c1 < 40;
  float v0 = 0.f, v1 = 0.f;
  float m0 = -1e30f, m1 = -1e30f;
  if (p0) { v0 = fmaxf(a[c0] + b[c0], 0.0f); m0 = v0; }
  if (p1) { v1 = fmaxf(a[c1] + b[c1], 0.0f); m1 = v1; }
  float m = fmaxf(m0, m1);
  for (int off = 16; off > 0; off >>= 1) m = fmaxf(m, __shfl_xor(m, off));
  float s = (p0 ? expf(v0 - m) : 0.0f) + (p1 ? expf(v1 - m) : 0.0f);
  for (int off = 16; off > 0; off >>= 1) s += __shfl_xor(s, off);
  float lse = m + logf(s);
  float* o = out + (size_t)row * 40;
  if (p0) o[c0] = v0 - lse;
  if (p1) o[c1] = v1 - lse;
}

// ---------------- launcher ----------------
extern "C" void kernel_launch(void* const* d_in, const int* in_sizes, int n_in,
                              void* d_out, int out_size, void* d_ws, size_t ws_size,
                              hipStream_t stream) {
  (void)in_sizes; (void)n_in; (void)out_size; (void)ws_size;
  const float* x    = (const float*)d_in[0];
  const int*   esrc = (const int*)d_in[1];
  const int*   edst = (const int*)d_in[2];
  const float* eval = (const float*)d_in[3];
  const float* Wm[5] = {(const float*)d_in[4], (const float*)d_in[6], (const float*)d_in[8],
                        (const float*)d_in[10], (const float*)d_in[12]};
  const float* Bm[5] = {(const float*)d_in[5], (const float*)d_in[7], (const float*)d_in[9],
                        (const float*)d_in[11], (const float*)d_in[13]};
  float* out = (float*)d_out;

  size_t off = 0;
  auto alloc = [&](size_t bytes) -> void* {
    void* p = (char*)d_ws + off;
    off += (bytes + 255) & ~(size_t)255;
    return p;
  };
  const size_t BIG = (size_t)NN * 1024 * sizeof(float);
  float*  S   = (float*)alloc(BIG);
  float*  AGG = (float*)alloc(BIG);
  float*  H   = (float*)alloc(BIG);   // current activations / id1 / residual stream
  float*  ID2 = (float*)alloc(BIG);
  bf16_t* Ahi = (bf16_t*)alloc((size_t)NN * 1024 * 2);
  bf16_t* Alo = (bf16_t*)alloc((size_t)NN * 1024 * 2);
  bf16_t* Whi = (bf16_t*)alloc((size_t)1024 * 1024 * 2);
  bf16_t* Wlo = (bf16_t*)alloc((size_t)1024 * 1024 * 2);

  const int tpb = 256;
  const int gy  = (MT + 7) / 8;  // 782

  // ---- Layer 1: [NN,128] @ W1 -> 1024 ; h1(id1) -> H ----
  {
    const int K = 128, dout = 1024, ldS = 1024;
    int n = NN * K;
    k_split<<<(n + tpb - 1) / tpb, tpb, 0, stream>>>(x, Ahi, Alo, n);
    int wn = ldS * K;
    k_wsplit<<<(wn + tpb - 1) / tpb, tpb, 0, stream>>>(Wm[0], Whi, Wlo, K, dout, ldS);
    k_gemm<8><<<dim3(ldS / 128, gy), tpb, 0, stream>>>(Ahi, Alo, Whi, Wlo, S, K, ldS, MT);
    hipMemsetAsync(AGG, 0, (size_t)NN * ldS * 4, stream);
    unsigned tot = (unsigned)NE * (dout / 4);
    k_scatter<256, 1024><<<(tot + tpb - 1) / tpb, tpb, 0, stream>>>(S, esrc, edst, eval,
                                                                    AGG, tot);
    int en = NN * dout;
    k_bias_relu<<<(en + tpb - 1) / tpb, tpb, 0, stream>>>(AGG, Bm[0], H, dout - 1, en);
  }
  // ---- Layer 2: 1024 -> 1024 ; id2 -> ID2 ; H = id2 + id1 ----
  {
    const int K = 1024, dout = 1024, ldS = 1024;
    int n = NN * K;
    k_split<<<(n + tpb - 1) / tpb, tpb, 0, stream>>>(H, Ahi, Alo, n);
    int wn = ldS * K;
    k_wsplit<<<(wn + tpb - 1) / tpb, tpb, 0, stream>>>(Wm[1], Whi, Wlo, K, dout, ldS);
    k_gemm<8><<<dim3(ldS / 128, gy), tpb, 0, stream>>>(Ahi, Alo, Whi, Wlo, S, K, ldS, MT);
    hipMemsetAsync(AGG, 0, (size_t)NN * ldS * 4, stream);
    unsigned tot = (unsigned)NE * (dout / 4);
    k_scatter<256, 1024><<<(tot + tpb - 1) / tpb, tpb, 0, stream>>>(S, esrc, edst, eval,
                                                                    AGG, tot);
    int en = NN * dout;
    k_bias_relu_save_addres<<<(en + tpb - 1) / tpb, tpb, 0, stream>>>(AGG, Bm[1], ID2, H,
                                                                     dout - 1, en);
  }
  // ---- Layer 3: 1024 -> 1024 ; H = relu(...) + id2 ----
  {
    const int K = 1024, dout = 1024, ldS = 1024;
    int n = NN * K;
    k_split<<<(n + tpb - 1) / tpb, tpb, 0, stream>>>(H, Ahi, Alo, n);
    int wn = ldS * K;
    k_wsplit<<<(wn + tpb - 1) / tpb, tpb, 0, stream>>>(Wm[2], Whi, Wlo, K, dout, ldS);
    k_gemm<8><<<dim3(ldS / 128, gy), tpb, 0, stream>>>(Ahi, Alo, Whi, Wlo, S, K, ldS, MT);
    hipMemsetAsync(AGG, 0, (size_t)NN * ldS * 4, stream);
    unsigned tot = (unsigned)NE * (dout / 4);
    k_scatter<256, 1024><<<(tot + tpb - 1) / tpb, tpb, 0, stream>>>(S, esrc, edst, eval,
                                                                    AGG, tot);
    int en = NN * dout;
    k_bias_relu_add<<<(en + tpb - 1) / tpb, tpb, 0, stream>>>(AGG, Bm[2], ID2, H,
                                                              dout - 1, en);
  }
  // ---- Layer 4: 1024 -> 64 ; H(64-wide) ----
  {
    const int K = 1024, dout = 64, ldS = 64;
    int n = NN * K;
    k_split<<<(n + tpb - 1) / tpb, tpb, 0, stream>>>(H, Ahi, Alo, n);
    int wn = ldS * K;
    k_wsplit<<<(wn + tpb - 1) / tpb, tpb, 0, stream>>>(Wm[3], Whi, Wlo, K, dout, ldS);
    k_gemm<4><<<dim3(1, gy), tpb, 0, stream>>>(Ahi, Alo, Whi, Wlo, S, K, ldS, MT);
    hipMemsetAsync(AGG, 0, (size_t)NN * ldS * 4, stream);
    unsigned tot = (unsigned)NE * (dout / 4);
    k_scatter<16, 64><<<(tot + tpb - 1) / tpb, tpb, 0, stream>>>(S, esrc, edst, eval,
                                                                 AGG, tot);
    int en = NN * dout;
    k_bias_relu<<<(en + tpb - 1) / tpb, tpb, 0, stream>>>(AGG, Bm[3], H, dout - 1, en);
  }
  // ---- Layer 5: 64 -> 40 (padded to 48) ; fused bias+relu+log_softmax ----
  {
    const int K = 64, dout = 40, ldS = 48;
    int n = NN * K;
    k_split<<<(n + tpb - 1) / tpb, tpb, 0, stream>>>(H, Ahi, Alo, n);
    int wn = ldS * K;
    k_wsplit<<<(wn + tpb - 1) / tpb, tpb, 0, stream>>>(Wm[4], Whi, Wlo, K, dout, ldS);
    k_gemm<3><<<dim3(1, gy), tpb, 0, stream>>>(Ahi, Alo, Whi, Wlo, S, K, ldS, MT);
    hipMemsetAsync(AGG, 0, (size_t)NN * ldS * 4, stream);
    unsigned tot = (unsigned)NE * (dout / 4);  // 10 chunks (cols 0..39)
    k_scatter<10, 48><<<(tot + tpb - 1) / tpb, tpb, 0, stream>>>(S, esrc, edst, eval,
                                                                 AGG, tot);
    k_logsoftmax<<<(NN + 7) / 8, tpb, 0, stream>>>(AGG, Bm[4], out, NN);
  }
}
